// ShiftedWindowAttention_26671746908306
// MI455X (gfx1250) — matrix operand order repaired and checked by reading, not compiled, not run on hardware
//
#include <hip/hip_runtime.h>

typedef __attribute__((ext_vector_type(16))) _Float16 v16h;
typedef __attribute__((ext_vector_type(8)))  float    v8f;

#define CC    192     // hidden
#define TT    64      // tokens per window
#define NHH   6       // heads
#define HDD   32      // head dim
#define WSTR  200     // LDS stride for win/O buffer (f16), 400B rows (16B aligned)
#define QSTR  40      // LDS stride for q/k      (f16),  80B rows
#define VSTR  72      // LDS stride for vT       (f16), 144B rows
#define PSTR  72      // LDS stride for probs    (f16), 144B rows
#define NQKV  (576*192)
#define NPROJ (192*192)
#define QSCALE 0.17677669529663687f   // 1/sqrt(32)

// ---- WMMA fragment loaders (ISA 7.12.2 16-bit layouts) ----
// A (16x32, row-major src): lane r=lane%16 -> row; hf=lane/16;
//   VGPR 0..3 : K = k0 + 8*hf + {0..7}   (contiguous 16B)
//   VGPR 4..7 : K = k0 + 16 + 8*hf + {0..7}
__device__ __forceinline__ v16h load_frag_a(const _Float16* base, int stride,
                                            int row0, int k0, int lane) {
  int r = lane & 15, hf = lane >> 4;
  const _Float16* p = base + (row0 + r) * stride + k0 + 8 * hf;
  union { v16h v; uint4 q[2]; } f;
  f.q[0] = *(const uint4*)p;
  f.q[1] = *(const uint4*)(p + 16);
  return f.v;
}
// B (32x16): source stored row-major [Ncols][K]; lane n=lane%16 -> column;
//   VGPR 0..3 : K = k0 + 8*hf + {0..7} ; VGPR 4..7 : K = k0 + 16 + 8*hf + {0..7}
__device__ __forceinline__ v16h load_frag_b(const _Float16* base, int stride,
                                            int col0, int k0, int lane) {
  int n = lane & 15, hf = lane >> 4;
  const _Float16* p = base + (col0 + n) * stride + k0 + 8 * hf;
  union { v16h v; uint4 q[2]; } f;
  f.q[0] = *(const uint4*)p;
  f.q[1] = *(const uint4*)(p + 16);
  return f.v;
}

// f32 -> f16 weight conversion; 1/sqrt(hd) folded into the Q rows (rows 0..191)
__global__ void cvt_weights(const float* __restrict__ Wq, const float* __restrict__ Wp,
                            _Float16* __restrict__ WqH, _Float16* __restrict__ WpH) {
  int i = blockIdx.x * blockDim.x + threadIdx.x;
  if (i < NQKV) {
    float v = Wq[i];
    if (i < 192 * 192) v *= QSCALE;   // q-section rows
    WqH[i] = (_Float16)v;
  } else {
    int j = i - NQKV;
    if (j < NPROJ) WpH[j] = (_Float16)Wp[j];
  }
}

__global__ __launch_bounds__(512) void swin_attn_fused(
    const float*    __restrict__ x,
    const _Float16* __restrict__ WqkvH,
    const float*    __restrict__ bqkv,
    const _Float16* __restrict__ WprojH,
    const float*    __restrict__ bproj,
    float*          __restrict__ out)
{
  __shared__ __align__(16) _Float16 sWin[TT * WSTR];        // window tokens, later O
  __shared__ __align__(16) _Float16 sQ  [NHH * TT * QSTR];  // q (pre-scaled)
  __shared__ __align__(16) _Float16 sK  [NHH * TT * QSTR];  // k
  __shared__ __align__(16) _Float16 sVT [NHH * HDD * VSTR]; // v^T [head][d][token]
  __shared__ __align__(16) _Float16 sP  [NHH * TT * PSTR];  // scores / probs
  __shared__ int sRow[TT];                                  // global row base per token

  const int tid  = threadIdx.x;
  const int lane = tid & 31;
  // wave id as a true scalar (SGPR) so tile indices become uniform for the compiler
  const int wvs  = __builtin_amdgcn_readfirstlane(tid >> 5);   // 0..15
  const int ln15 = lane & 15;
  const int hf   = lane >> 4;

  // --- decode window: (b, hw, ww, dw), 12 windows per dim ---
  int w  = blockIdx.x;
  int b  = w / 1728;
  int r3 = w % 1728;
  int hw = r3 / 144;
  int ww = (r3 / 12) % 12;
  int dw = r3 % 12;

  // shift roll: token (i,j,k) <- x at ((base+i+2)%48, ...); same index for scatter
  if (tid < TT) {
    int t = tid;
    int i = t >> 4, j = (t >> 2) & 3, k = t & 3;
    int sh = (hw * 4 + i + 2) % 48;
    int sw = (ww * 4 + j + 2) % 48;
    int sd = (dw * 4 + k + 2) % 48;
    sRow[t] = (b * 110592 + (sh * 48 + sw) * 48 + sd) * CC;
  }
  __syncthreads();

  // --- phase 1: gather window (64 x 192 f32) -> f16 LDS, packed b128 stores ---
  {
    int tok = tid >> 3;           // 8 threads per token
    int c0  = (tid & 7) * 24;     // 24 channels each
    const float4* src = (const float4*)(x + sRow[tok] + c0);
    uint4* dst = (uint4*)(sWin + tok * WSTR + c0);   // 16B aligned
    #pragma unroll
    for (int q = 0; q < 3; ++q) {
      float4 a = src[2 * q], c = src[2 * q + 1];
      union { uint4 u; _Float16 h[8]; } pk;
      pk.h[0] = (_Float16)a.x; pk.h[1] = (_Float16)a.y;
      pk.h[2] = (_Float16)a.z; pk.h[3] = (_Float16)a.w;
      pk.h[4] = (_Float16)c.x; pk.h[5] = (_Float16)c.y;
      pk.h[6] = (_Float16)c.z; pk.h[7] = (_Float16)c.w;
      dst[q] = pk.u;
    }
  }
  __syncthreads();

  // --- phase 2: QKV GEMM (64x192 @ 192x576) ---
  for (int tile = wvs; tile < 144; tile += 16) {   // 4 mtiles x 36 ntiles (scalar)
    int mt = tile & 3;
    int nt = tile >> 2;
    v8f acc = {};
    #pragma unroll
    for (int ks = 0; ks < 6; ++ks) {
      v16h A = load_frag_a(sWin, WSTR, mt * 16, ks * 32, lane);
      v16h B = load_frag_b(WqkvH, CC, nt * 16, ks * 32, lane);
      acc = __builtin_amdgcn_wmma_f32_16x16x32_f16(false, A, false, B,
                                                   (short)0, acc, false, false);
    }
    // all of sec/head/ntl are SGPR values -> scalar branches, SALU address math
    int sec   = nt / 12;            // 0=q 1=k 2=v
    int ntl   = nt - sec * 12;
    int head  = ntl >> 1;
    int dbase = (ntl & 1) << 4;
    int d     = dbase + ln15;
    float bv  = bqkv[nt * 16 + ln15];
    if (sec == 0) bv *= QSCALE;     // weights pre-scaled; scale bias to match
    int m0    = mt * 16 + 8 * hf;
    if (sec == 0) {
      _Float16* q = &sQ[(head * TT + m0) * QSTR + d];
      #pragma unroll
      for (int j = 0; j < 8; ++j) q[j * QSTR] = (_Float16)(acc[j] + bv);
    } else if (sec == 1) {
      _Float16* kk = &sK[(head * TT + m0) * QSTR + d];
      #pragma unroll
      for (int j = 0; j < 8; ++j) kk[j * QSTR] = (_Float16)(acc[j] + bv);
    } else {
      // v^T: per lane, 8 consecutive tokens -> one b128 store
      union { uint4 u; _Float16 h[8]; } pk;
      #pragma unroll
      for (int j = 0; j < 8; ++j) pk.h[j] = (_Float16)(acc[j] + bv);
      *(uint4*)&sVT[(head * HDD + d) * VSTR + m0] = pk.u;   // 16B aligned
    }
  }
  __syncthreads();

  // --- phase 3: scores P = qs @ k^T (per head 64x32 @ 32x64, K=32 = 1 WMMA) ---
  for (int tile = wvs; tile < 96; tile += 16) {    // 6 heads x 4 mt x 4 nt
    int h  = tile >> 4;
    int mt = (tile >> 2) & 3;
    int nt = tile & 3;
    v16h A = load_frag_a(sQ + h * TT * QSTR, QSTR, mt * 16, 0, lane);
    v16h B = load_frag_b(sK + h * TT * QSTR, QSTR, nt * 16, 0, lane);
    v8f acc = {};
    acc = __builtin_amdgcn_wmma_f32_16x16x32_f16(false, A, false, B,
                                                 (short)0, acc, false, false);
    _Float16* p = &sP[(h * TT + mt * 16 + 8 * hf) * PSTR + nt * 16 + ln15];
    #pragma unroll
    for (int j = 0; j < 8; ++j) p[j * PSTR] = (_Float16)acc[j];
  }
  __syncthreads();

  // --- phase 4: softmax over 384 rows of 64, vectorized b128 LDS traffic ---
  if (tid < NHH * TT) {
    uint4* row4 = (uint4*)(sP + tid * PSTR);     // 144B row stride, 16B aligned
    union { uint4 u; _Float16 h[8]; } c[8];
    float mx = -1e30f;
    #pragma unroll
    for (int i = 0; i < 8; ++i) {
      c[i].u = row4[i];
      #pragma unroll
      for (int j = 0; j < 8; ++j) mx = fmaxf(mx, (float)c[i].h[j]);
    }
    float s = 0.f;
    #pragma unroll
    for (int i = 0; i < 8; ++i)
      #pragma unroll
      for (int j = 0; j < 8; ++j) {
        float e = __expf((float)c[i].h[j] - mx);
        s += e;
        c[i].h[j] = (_Float16)e;
      }
    float inv = 1.f / s;
    #pragma unroll
    for (int i = 0; i < 8; ++i) {
      #pragma unroll
      for (int j = 0; j < 8; ++j)
        c[i].h[j] = (_Float16)((float)c[i].h[j] * inv);
      row4[i] = c[i].u;
    }
  }
  __syncthreads();

  // --- phase 5: O = P @ V (per head 64x64 @ 64x32, K=64 = 2 steps), O -> sWin ---
  for (int tile = wvs; tile < 48; tile += 16) {    // 6 heads x 4 mt x 2 nt
    int h  = tile >> 3;
    int mt = (tile >> 1) & 3;
    int nt = tile & 1;
    v8f acc = {};
    #pragma unroll
    for (int ks = 0; ks < 2; ++ks) {
      v16h A = load_frag_a(sP + h * TT * PSTR, PSTR, mt * 16, ks * 32, lane);
      v16h B = load_frag_b(sVT + h * HDD * VSTR, VSTR, nt * 16, ks * 32, lane);
      acc = __builtin_amdgcn_wmma_f32_16x16x32_f16(false, A, false, B,
                                                   (short)0, acc, false, false);
    }
    _Float16* o = &sWin[(mt * 16 + 8 * hf) * WSTR + h * HDD + nt * 16 + ln15];
    #pragma unroll
    for (int j = 0; j < 8; ++j) o[j * WSTR] = (_Float16)acc[j];
  }
  __syncthreads();

  // --- phase 6: proj (64x192 @ 192x192) + bias + scatter with reverse shift ---
  for (int tile = wvs; tile < 48; tile += 16) {    // 4 mt x 12 nt
    int mt = tile & 3;
    int nt = tile >> 2;
    v8f acc = {};
    #pragma unroll
    for (int ks = 0; ks < 6; ++ks) {
      v16h A = load_frag_a(sWin, WSTR, mt * 16, ks * 32, lane);
      v16h B = load_frag_b(WprojH, CC, nt * 16, ks * 32, lane);
      acc = __builtin_amdgcn_wmma_f32_16x16x32_f16(false, A, false, B,
                                                   (short)0, acc, false, false);
    }
    int n    = nt * 16 + ln15;
    float bv = bproj[n];
    int m0   = mt * 16 + 8 * hf;
    #pragma unroll
    for (int j = 0; j < 8; ++j) {
      out[sRow[m0 + j] + n] = acc[j] + bv;
    }
  }
}

extern "C" void kernel_launch(void* const* d_in, const int* in_sizes, int n_in,
                              void* d_out, int out_size, void* d_ws, size_t ws_size,
                              hipStream_t stream) {
  const float* x     = (const float*)d_in[0];
  const float* Wqkv  = (const float*)d_in[1];
  const float* bqkv  = (const float*)d_in[2];
  const float* Wproj = (const float*)d_in[3];
  const float* bproj = (const float*)d_in[4];
  float* out = (float*)d_out;

  _Float16* WqkvH  = (_Float16*)d_ws;
  _Float16* WprojH = WqkvH + NQKV;   // 16B-aligned offset (221184 B)

  // convert weights to f16 once per launch (147456 elements total)
  cvt_weights<<<(NQKV + NPROJ + 511) / 512, 512, 0, stream>>>(Wqkv, Wproj, WqkvH, WprojH);

  // one workgroup per window: 2 * 12^3 = 3456 windows, 16 waves each
  swin_attn_fused<<<3456, 512, 0, stream>>>(x, WqkvH, bqkv, WprojH, bproj, out);
}
